// DynamicBlockSparseMoE_10952166604908
// MI455X (gfx1250) — matrix operand order repaired and checked by reading, not compiled, not run on hardware
//
#include <hip/hip_runtime.h>

// MoE block-sparse forward, MI455X (gfx1250, wave32).
// gating -> colsum(x)@gate_w -> top-4; pre-convert/transpose selected blocks to
// bf16 once; two 4096^3 bf16 WMMA GEMMs (f32 accum) with double-buffered LDS
// and async global->LDS copies (GLOBAL_LOAD_ASYNC_TO_LDS_B128 / ASYNCcnt).
// Workspace: 256KB partials | colsum | sel | xb(32MB) | Wt(32MB) | At2(32MB) | H(32MB)
// => needs ~130MB of d_ws.

typedef __attribute__((ext_vector_type(16))) __bf16 v16bf;
typedef __attribute__((ext_vector_type(8)))  __bf16 v8bf;
typedef __attribute__((ext_vector_type(4)))  __bf16 v4bf;
typedef __attribute__((ext_vector_type(8)))  float  v8f;
typedef __attribute__((ext_vector_type(4)))  int    v4i;

// address-space qualified element types for the async-copy builtin
typedef __attribute__((address_space(1))) v4i as1_v4i;
typedef __attribute__((address_space(3))) v4i as3_v4i;

#define DIM  4096   // BATCH == INPUT_DIM == OUTPUT_DIM == TOPK*NHID
#define NEXP 16
#define NHID 1024
#define TOPK 4

#if defined(__has_builtin)
#if __has_builtin(__builtin_amdgcn_global_load_async_to_lds_b128)
#define HAVE_ASYNC_COPY 1
#endif
#endif
#ifndef HAVE_ASYNC_COPY
#define HAVE_ASYNC_COPY 0
#pragma message("gfx1250 async-to-LDS builtin not found; using VGPR-staged double buffering")
#endif

#if HAVE_ASYNC_COPY
static __device__ __forceinline__ void wait_async0() {
#if __has_builtin(__builtin_amdgcn_s_wait_asynccnt)
  __builtin_amdgcn_s_wait_asynccnt(0);
#else
  asm volatile("s_wait_asynccnt 0x0" ::: "memory");
#endif
}
#endif

// ---------- gating: partial column sums of x ----------
__global__ __launch_bounds__(256) void partial_colsum_kernel(
    const float* __restrict__ x, float* __restrict__ partial) {
  int col  = blockIdx.x * 256 + threadIdx.x;
  int row0 = blockIdx.y * 256;
  float s = 0.f;
  #pragma unroll 4
  for (int r = 0; r < 256; ++r)
    s += x[(size_t)(row0 + r) * DIM + col];
  partial[(size_t)blockIdx.y * DIM + col] = s;
}

__global__ __launch_bounds__(256) void reduce_colsum_kernel(
    const float* __restrict__ partial, float* __restrict__ colsum) {
  int col = blockIdx.x * 256 + threadIdx.x;
  float s = 0.f;
  #pragma unroll
  for (int j = 0; j < 16; ++j) s += partial[(size_t)j * DIM + col];
  colsum[col] = s;
}

// ---------- gating matvec + top-4 ----------
__global__ __launch_bounds__(512) void gate_topk_kernel(
    const float* __restrict__ colsum, const float* __restrict__ gw,
    const float* __restrict__ gb, int* __restrict__ sel) {
  __shared__ float gs[NEXP];
  int tid = threadIdx.x;
  int e = tid >> 5, lane = tid & 31;            // one wave32 per expert
  float s = 0.f;
  for (int k = lane; k < DIM; k += 32)
    s += colsum[k] * gw[(size_t)k * NEXP + e];
  #pragma unroll
  for (int off = 16; off > 0; off >>= 1) s += __shfl_down(s, off);
  if (lane == 0) gs[e] = s + (float)DIM * gb[e];
  __syncthreads();
  if (tid == 0) {
    bool used[NEXP];
    #pragma unroll
    for (int i = 0; i < NEXP; ++i) used[i] = false;
    for (int t = 0; t < TOPK; ++t) {
      int best = 0; float bv = -3.402823466e38f;
      for (int i = 0; i < NEXP; ++i)
        if (!used[i] && gs[i] > bv) { bv = gs[i]; best = i; }
      used[best] = true;
      sel[t] = best;
    }
  }
}

// ---------- f32 -> bf16 bulk convert (x) ----------
__global__ __launch_bounds__(256) void cvt_bf16_kernel(
    const float* __restrict__ src, __bf16* __restrict__ dst) {
  size_t i = ((size_t)blockIdx.x * 256 + threadIdx.x) * 4;
  float4 v = *(const float4*)(src + i);
  v4bf b;
  b[0] = (__bf16)v.x; b[1] = (__bf16)v.y; b[2] = (__bf16)v.z; b[3] = (__bf16)v.w;
  *(v4bf*)(dst + i) = b;
}

// ---------- select + transpose + convert: dst[n][k] (bf16, ld=DIM) ----------
// ROWSEL=0 (weight): dst[n][k] = src[k][selcol(n)]
// ROWSEL=1 (agg_w) : dst[n][k] = src[selrow(k)][n]
template<bool ROWSEL>
__global__ __launch_bounds__(256) void transpose_sel_kernel(
    const float* __restrict__ src, __bf16* __restrict__ dst,
    const int* __restrict__ selp, int ldsrc) {
  __shared__ float t[32][33];
  const int tx = threadIdx.x, ty = threadIdx.y;
  const int n0 = blockIdx.x * 32, k0 = blockIdx.y * 32;
  int se[TOPK];
  #pragma unroll
  for (int i = 0; i < TOPK; ++i) se[i] = selp[i];
  #pragma unroll
  for (int i = 0; i < 4; ++i) {
    int k = k0 + ty + 8 * i;
    int n = n0 + tx;
    int sr, sc;
    if (ROWSEL) { sr = (se[k >> 10] << 10) | (k & (NHID - 1)); sc = n; }
    else        { sr = k; sc = (se[n >> 10] << 10) | (n & (NHID - 1)); }
    t[ty + 8 * i][tx] = src[(size_t)sr * ldsrc + sc];
  }
  __syncthreads();
  #pragma unroll
  for (int i = 0; i < 4; ++i) {
    int n = n0 + ty + 8 * i;
    int k = k0 + tx;
    dst[(size_t)n * DIM + k] = (__bf16)t[tx][ty + 8 * i];
  }
}

// ---------- bf16 WMMA GEMM: C[M][N] = A[M][K] @ Bt[N][K]^T ----------
template<bool CBF16, bool BIAS>
__global__ __launch_bounds__(256) void moe_gemm_kernel(
    const __bf16* __restrict__ A,   // [M][K] row-major bf16
    const __bf16* __restrict__ Bt,  // [N][K] row-major bf16 (pre-transposed, pre-selected)
    const float* __restrict__ bias,
    void* __restrict__ Cv, int K, int lda, int ldb, int ldc) {
  constexpr int BM = 128, BN = 128, BK = 32;
  constexpr int LA = 40, LB = 40;                 // 80B row stride: frag loads conflict-free
  __shared__ __attribute__((aligned(16))) __bf16 As[2][BM * LA];
  __shared__ __attribute__((aligned(16))) __bf16 Bs[2][BN * LB];

  const int tid  = threadIdx.x;
  const int lane = tid & 31;
  const int wave = tid >> 5;                      // 8 waves: 4 (m) x 2 (n)
  const int wm   = wave & 3;
  const int wn   = wave >> 2;
  const int half = lane >> 4;
  const int l16  = lane & 15;
  const int rowBase = blockIdx.y * BM;
  const int colBase = blockIdx.x * BN;

  v8f acc[2][4];
  #pragma unroll
  for (int mt = 0; mt < 2; ++mt)
    #pragma unroll
    for (int nt = 0; nt < 4; ++nt)
      #pragma unroll
      for (int r = 0; r < 8; ++r) acc[mt][nt][r] = 0.f;

  auto compute = [&](int buf) {
    v16bf aF[2], bF[4];
    #pragma unroll
    for (int mt = 0; mt < 2; ++mt) {
      const __bf16* rp = &As[buf][(wm * 32 + mt * 16 + l16) * LA + half * 8];
      v8bf lo = *(const v8bf*)rp;
      v8bf hi = *(const v8bf*)(rp + 16);
      #pragma unroll
      for (int i = 0; i < 8; ++i) { aF[mt][i] = lo[i]; aF[mt][i + 8] = hi[i]; }
    }
    #pragma unroll
    for (int nt = 0; nt < 4; ++nt) {
      const __bf16* cp = &Bs[buf][(wn * 64 + nt * 16 + l16) * LB + half * 8];
      v8bf lo = *(const v8bf*)cp;
      v8bf hi = *(const v8bf*)(cp + 16);
      #pragma unroll
      for (int i = 0; i < 8; ++i) { bF[nt][i] = lo[i]; bF[nt][i + 8] = hi[i]; }
    }
    #pragma unroll
    for (int mt = 0; mt < 2; ++mt)
      #pragma unroll
      for (int nt = 0; nt < 4; ++nt)
        acc[mt][nt] = __builtin_amdgcn_wmma_f32_16x16x32_bf16(
            false, aF[mt], false, bF[nt], (short)0, acc[mt][nt], false, false);
  };

#if HAVE_ASYNC_COPY
  // 128x32 bf16 tile = 512 x 16B chunks; 2 per thread per matrix, fully async.
  auto issueTile = [&](int kt, int buf) {
    #pragma unroll
    for (int j = 0; j < 2; ++j) {
      int pos = tid + j * 256;
      int r  = pos >> 2;              // 4 chunks per row
      int c8 = (pos & 3) << 3;
      __builtin_amdgcn_global_load_async_to_lds_b128(
          (as1_v4i*)(A + (size_t)(rowBase + r) * lda + kt + c8),
          (as3_v4i*)(&As[buf][r * LA + c8]), 0, 0);
      __builtin_amdgcn_global_load_async_to_lds_b128(
          (as1_v4i*)(Bt + (size_t)(colBase + r) * ldb + kt + c8),
          (as3_v4i*)(&Bs[buf][r * LB + c8]), 0, 0);
    }
  };
  issueTile(0, 0);
  wait_async0();
  int p = 0;
  for (int kt = 0; kt < K; kt += BK) {
    bool more = (kt + BK) < K;
    __syncthreads();                   // buf 1-p now free (read last iteration)
    if (more) issueTile(kt + BK, 1 - p);
    compute(p);
    if (more) wait_async0();           // own async writes done before next barrier
    p ^= 1;
  }
#else
  v8bf aReg[2], bReg[2];
  auto loadTile = [&](int kt) {
    #pragma unroll
    for (int j = 0; j < 2; ++j) {
      int pos = tid + j * 256;
      int r  = pos >> 2;
      int c8 = (pos & 3) << 3;
      aReg[j] = *(const v8bf*)(A + (size_t)(rowBase + r) * lda + kt + c8);
      bReg[j] = *(const v8bf*)(Bt + (size_t)(colBase + r) * ldb + kt + c8);
    }
  };
  auto storeTile = [&](int buf) {
    #pragma unroll
    for (int j = 0; j < 2; ++j) {
      int pos = tid + j * 256;
      int r  = pos >> 2;
      int c8 = (pos & 3) << 3;
      *(v8bf*)&As[buf][r * LA + c8] = aReg[j];
      *(v8bf*)&Bs[buf][r * LB + c8] = bReg[j];
    }
  };
  loadTile(0);
  storeTile(0);
  int p = 0;
  for (int kt = 0; kt < K; kt += BK) {
    bool more = (kt + BK) < K;
    if (more) loadTile(kt + BK);       // global->regs overlaps compute
    __syncthreads();
    compute(p);
    if (more) storeTile(1 - p);
    p ^= 1;
  }
#endif

  // epilogue: D layout VGPR r -> M = r + 8*half, N = l16
  #pragma unroll
  for (int mt = 0; mt < 2; ++mt) {
    #pragma unroll
    for (int nt = 0; nt < 4; ++nt) {
      int row = rowBase + wm * 32 + mt * 16 + half * 8;
      int col = colBase + wn * 64 + nt * 16 + l16;
      if constexpr (CBF16) {
        __bf16* Cb = (__bf16*)Cv;
        #pragma unroll
        for (int r = 0; r < 8; ++r)
          Cb[(size_t)(row + r) * ldc + col] = (__bf16)acc[mt][nt][r];
      } else {
        float* Cf = (float*)Cv;
        float bv = BIAS ? bias[col] : 0.f;
        #pragma unroll
        for (int r = 0; r < 8; ++r)
          Cf[(size_t)(row + r) * ldc + col] = BIAS ? (acc[mt][nt][r] + bv)
                                                   : acc[mt][nt][r];
      }
    }
  }
}

extern "C" void kernel_launch(void* const* d_in, const int* in_sizes, int n_in,
                              void* d_out, int out_size, void* d_ws, size_t ws_size,
                              hipStream_t stream) {
  const float* x      = (const float*)d_in[0];
  const float* gate_w = (const float*)d_in[1];
  const float* gate_b = (const float*)d_in[2];
  const float* weight = (const float*)d_in[3];
  const float* agg_w  = (const float*)d_in[4];
  const float* agg_b  = (const float*)d_in[5];
  float* out = (float*)d_out;

  char* ws = (char*)d_ws;
  const size_t MB = 1u << 20;
  float*  partial = (float*)ws;                          // 16*4096 f32 = 256 KB
  float*  colsum  = (float*)(ws + 262144);               // 16 KB
  int*    sel     = (int*)(ws + 262144 + 16384);
  __bf16* xb      = (__bf16*)(ws + 1 * MB);              // 32 MB  x as bf16
  __bf16* Wt      = (__bf16*)(ws + 33 * MB);             // 32 MB  weight selected, [n][k]
  __bf16* At2     = (__bf16*)(ws + 65 * MB);             // 32 MB  agg_w selected, [n][k]
  __bf16* H       = (__bf16*)(ws + 97 * MB);             // 32 MB  intermediate

  // 1) gating (deterministic, no atomics)
  partial_colsum_kernel<<<dim3(16, 16), 256, 0, stream>>>(x, partial);
  reduce_colsum_kernel<<<dim3(16), 256, 0, stream>>>(partial, colsum);
  gate_topk_kernel<<<dim3(1), 512, 0, stream>>>(colsum, gate_w, gate_b, sel);

  // 2) one-time convert / select+transpose into bf16
  cvt_bf16_kernel<<<dim3(16384), 256, 0, stream>>>(x, xb);
  transpose_sel_kernel<false><<<dim3(128, 128), dim3(32, 8), 0, stream>>>(
      weight, Wt, sel, NEXP * NHID);
  transpose_sel_kernel<true><<<dim3(128, 128), dim3(32, 8), 0, stream>>>(
      agg_w, At2, sel, DIM);

  // 3) H = xb @ Wt^T   (bf16 out)
  moe_gemm_kernel<true, false><<<dim3(32, 32), 256, 0, stream>>>(
      xb, Wt, nullptr, H, DIM, DIM, DIM, DIM);
  // 4) out = H @ At2^T + agg_b   (f32 out)
  moe_gemm_kernel<false, true><<<dim3(32, 32), 256, 0, stream>>>(
      H, At2, agg_b, out, DIM, DIM, DIM, DIM);
}
